// GATLayer_9904194584942
// MI455X (gfx1250) — matrix operand order repaired and checked by reading, not compiled, not run on hardware
//
#include <hip/hip_runtime.h>
#include <hip/hip_bf16.h>

typedef __attribute__((ext_vector_type(16))) _Float16 v16h;
typedef __attribute__((ext_vector_type(8)))  _Float16 v8h;
typedef __attribute__((ext_vector_type(8)))  float    v8f;

#define NROWS 16384
#define DIM   128
#define SCALE 0.08838834764831845f   // 1/sqrt(128)
#define KTILE 32
#define NTILES (NROWS / KTILE)       // 512

// ---- helpers -------------------------------------------------------------

__device__ __forceinline__ v16h ld16h(const _Float16* p) {
  v8h a = *(const v8h*)(p);
  v8h b = *(const v8h*)(p + 8);
  v16h r;
#pragma unroll
  for (int i = 0; i < 8; ++i) { r[i] = a[i]; r[8 + i] = b[i]; }
  return r;
}

__device__ __forceinline__ v8f wmma_f16(v16h a, v16h b, v8f c) {
  return __builtin_amdgcn_wmma_f32_16x16x32_f16(false, a, false, b,
                                                (short)0, c, false, false);
}

// CDNA5 async copy: global -> LDS, 16B per lane, tracked by ASYNCcnt.
__device__ __forceinline__ void async_b128(const void* gptr, void* lptr) {
  const unsigned lds = (unsigned)(uintptr_t)lptr;           // LDS byte offset
  const unsigned long long ga = (unsigned long long)(uintptr_t)gptr;
  asm volatile("global_load_async_to_lds_b128 %0, %1, off"
               :: "v"(lds), "v"(ga) : "memory");
}

__device__ __forceinline__ void wait_async0() {
#if __has_builtin(__builtin_amdgcn_s_wait_asynccnt)
  __builtin_amdgcn_s_wait_asynccnt(0);
#else
  asm volatile("s_wait_asynccnt 0" ::: "memory");
#endif
}

// ---- kernel 1: q/k/v projections (f16 outputs, q pre-scaled, v transposed)

__device__ __forceinline__ void proj_rowmajor(const v16h qa[4], const float* W,
                                              const float* bias, float s,
                                              _Float16* out, int r0, int lane) {
  const int c16 = lane & 15, hl = lane >> 4;
  const int klo16 = 16 * hl;
#pragma unroll
  for (int dt = 0; dt < 8; ++dt) {
    const int col = dt * 16 + c16;
    v8f acc = {};
#pragma unroll
    for (int kb = 0; kb < 4; ++kb) {
      const float* wp = W + (size_t)col * DIM + kb * 32 + klo16;
      v16h bw;
#pragma unroll
      for (int i = 0; i < 16; ++i) bw[i] = (_Float16)wp[i];
      acc = wmma_f16(qa[kb], bw, acc);
    }
    const float bb = bias[col];
#pragma unroll
    for (int r = 0; r < 8; ++r)
      out[(size_t)(r0 + r + 8 * hl) * DIM + col] = (_Float16)((acc[r] + bb) * s);
  }
}

__device__ __forceinline__ void proj_transposed(const v16h qa[4], const float* W,
                                                const float* bias,
                                                _Float16* outT, int r0, int lane) {
  const int c16 = lane & 15, hl = lane >> 4;
  const int klo16 = 16 * hl;
#pragma unroll
  for (int dt = 0; dt < 8; ++dt) {
    const int col = dt * 16 + c16;
    v8f acc = {};
#pragma unroll
    for (int kb = 0; kb < 4; ++kb) {
      const float* wp = W + (size_t)col * DIM + kb * 32 + klo16;
      v16h bw;
#pragma unroll
      for (int i = 0; i < 16; ++i) bw[i] = (_Float16)wp[i];
      acc = wmma_f16(qa[kb], bw, acc);
    }
    const float bb = bias[col];
    v8h pk;
#pragma unroll
    for (int r = 0; r < 8; ++r) pk[r] = (_Float16)(acc[r] + bb);
    // Vt[col][r0 + 8*hl + 0..7] : 16B contiguous store per lane
    *(v8h*)(outT + (size_t)col * NROWS + r0 + 8 * hl) = pk;
  }
}

__global__ __launch_bounds__(256) void qkv_kernel(
    const float* __restrict__ X,
    const float* __restrict__ Wq, const float* __restrict__ bq,
    const float* __restrict__ Wk, const float* __restrict__ bk,
    const float* __restrict__ Wv, const float* __restrict__ bv,
    _Float16* __restrict__ Qs, _Float16* __restrict__ Kh,
    _Float16* __restrict__ Vt) {
  const int lane = threadIdx.x & 31;
  const int wave = threadIdx.x >> 5;
  const int r0 = blockIdx.x * 128 + wave * 16;
  const int c16 = lane & 15, hl = lane >> 4;

  v16h xa[4];
  const float* xrow = X + (size_t)(r0 + c16) * DIM;
#pragma unroll
  for (int kb = 0; kb < 4; ++kb) {
    const float* p1 = xrow + kb * 32 + 16 * hl;
    const float* p2 = p1 + 8;
#pragma unroll
    for (int i = 0; i < 8; ++i) {
      xa[kb][i]     = (_Float16)p1[i];
      xa[kb][8 + i] = (_Float16)p2[i];
    }
  }
  proj_rowmajor (xa, Wq, bq, SCALE, Qs, r0, lane);  // q * (1/sqrt(D)) folded in
  proj_rowmajor (xa, Wk, bk, 1.0f,  Kh, r0, lane);
  proj_transposed(xa, Wv, bv,       Vt, r0, lane);
}

// ---- kernel 2: fused flash attention with zero-fill masking --------------

__global__ __launch_bounds__(256) void attn_kernel(
    const _Float16* __restrict__ Q, const _Float16* __restrict__ K,
    const _Float16* __restrict__ Vt, const int* __restrict__ adj,
    float* __restrict__ out) {
  __shared__ _Float16 sK[2][KTILE][136];  // padded: 68-dword row stride
  __shared__ _Float16 sV[2][DIM][40];     // padded: 20-dword row stride
  __shared__ _Float16 sP[8][16][40];      // per-wave P relayout tile

  const int tid  = threadIdx.x;
  const int lane = tid & 31;
  const int wave = tid >> 5;
  const int c16  = lane & 15;
  const int hl   = lane >> 4;
  const int qb   = blockIdx.x * 128 + wave * 16;

  // Q A-fragments for this wave's 16 rows (K = 0..127 in 4 chunks of 32)
  v16h qa[4];
  {
    const _Float16* qrow = Q + (size_t)(qb + c16) * DIM;
#pragma unroll
    for (int kb = 0; kb < 4; ++kb) {
      const _Float16* p1 = qrow + kb * 32 + 16 * hl;
      v8h a = *(const v8h*)p1;
      v8h b = *(const v8h*)(p1 + 8);
#pragma unroll
      for (int i = 0; i < 8; ++i) { qa[kb][i] = a[i]; qa[kb][8 + i] = b[i]; }
    }
  }

  // constant ones B-fragment: row-sum of P via one WMMA
  v16h onesB;
#pragma unroll
  for (int i = 0; i < 16; ++i) onesB[i] = (_Float16)1.0f;

  float m[8], l[8];
  v8f   O[8];
#pragma unroll
  for (int r = 0; r < 8; ++r) { m[r] = -1e30f; l[r] = 0.0f; }
#pragma unroll
  for (int t = 0; t < 8; ++t) O[t] = v8f{};

  // async tile staging (global -> LDS via ASYNCcnt, double buffered)
  const int krow = tid >> 3, kseg = tid & 7;  // K tile: 32 rows x 8 x 32B
  const int vrow = tid >> 1, vseg = tid & 1;  // Vt tile: 128 rows x 2 x 32B

  auto issueAsync = [&](int jb, int buf) {
    const _Float16* gk = K + (size_t)(jb + krow) * DIM + kseg * 16;
    _Float16* lk = &sK[buf][krow][kseg * 16];
    async_b128(gk, lk);
    async_b128(gk + 8, lk + 8);
    const _Float16* gv = Vt + (size_t)vrow * NROWS + jb + vseg * 16;
    _Float16* lv = &sV[buf][vrow][vseg * 16];
    async_b128(gv, lv);
    async_b128(gv + 8, lv + 8);
  };

  issueAsync(0, 0);

  for (int j = 0; j < NTILES; ++j) {
    wait_async0();       // this wave's async writes into buf (j&1) complete
    __syncthreads();     // all waves' writes visible; prior readers of buf^1 done
    const int jb = j * KTILE;
    if (j + 1 < NTILES) {
      issueAsync(jb + KTILE, (j + 1) & 1);
      // prefetch next adj tile (16 rows x 32 ints covered by 32 lanes)
      __builtin_prefetch(adj + (size_t)(qb + c16) * NROWS + jb + KTILE + 16 * hl, 0, 0);
    }
    const int buf = j & 1;

    // ---- S = q @ k^T (pre-scaled), zero-fill masked by adj ----
    float p0[8], p1[8];
#pragma unroll
    for (int sub = 0; sub < 2; ++sub) {
      v8f acc = {};
      const int srow = sub * 16 + c16;
#pragma unroll
      for (int kb = 0; kb < 4; ++kb) {
        v16h bK = ld16h(&sK[buf][srow][kb * 32 + 16 * hl]);
        acc = wmma_f16(qa[kb], bK, acc);
      }
      const int col = jb + sub * 16 + c16;
      float* dst = sub ? p1 : p0;
#pragma unroll
      for (int r = 0; r < 8; ++r) {
        const int a = __builtin_nontemporal_load(
            adj + (size_t)(qb + r + 8 * hl) * NROWS + col);
        dst[r] = a ? acc[r] : 0.0f;   // zero-fill (NOT -inf) per reference
      }
    }

    // ---- online softmax: row max (shuffles), conditional rescale ----
    float mn_[8];
    int chg = 0;
#pragma unroll
    for (int r = 0; r < 8; ++r) {
      float t = fmaxf(p0[r], p1[r]);
      t = fmaxf(t, __shfl_xor(t, 1, 32));
      t = fmaxf(t, __shfl_xor(t, 2, 32));
      t = fmaxf(t, __shfl_xor(t, 4, 32));
      t = fmaxf(t, __shfl_xor(t, 8, 32));
      mn_[r] = fmaxf(m[r], t);
      chg |= (mn_[r] > m[r]) ? 1 : 0;
    }
    if (__any(chg)) {          // rare after the running max stabilizes
#pragma unroll
      for (int r = 0; r < 8; ++r) {
        const float cc = __expf(m[r] - mn_[r]);
        m[r] = mn_[r];
        l[r] *= cc;
#pragma unroll
        for (int t = 0; t < 8; ++t) O[t][r] *= cc;
      }
    }
#pragma unroll
    for (int r = 0; r < 8; ++r) {
      p0[r] = __expf(p0[r] - m[r]);
      p1[r] = __expf(p1[r] - m[r]);
    }

    // ---- relayout P (C layout) -> A fragment via per-wave LDS tile ----
#pragma unroll
    for (int r = 0; r < 8; ++r) {
      sP[wave][r + 8 * hl][c16]      = (_Float16)p0[r];
      sP[wave][r + 8 * hl][16 + c16] = (_Float16)p1[r];
    }
    v16h pa;
    {
      const _Float16* pr = &sP[wave][c16][16 * hl];
      v8h a = *(const v8h*)pr;
      v8h b = *(const v8h*)(pr + 8);
#pragma unroll
      for (int i = 0; i < 8; ++i) { pa[i] = a[i]; pa[8 + i] = b[i]; }
    }

    // ---- row sums of the f16-rounded P via one WMMA (ones B-matrix) ----
    {
      v8f rs = wmma_f16(pa, onesB, v8f{});
#pragma unroll
      for (int r = 0; r < 8; ++r) l[r] += rs[r];
    }

    // ---- O += P @ V  (V columns come from transposed-V LDS tile) ----
#pragma unroll
    for (int dt = 0; dt < 8; ++dt) {
      v16h bV = ld16h(&sV[buf][dt * 16 + c16][16 * hl]);
      O[dt] = wmma_f16(pa, bV, O[dt]);
    }
  }

  // ---- epilogue: normalize and store ----
  float inv[8];
#pragma unroll
  for (int r = 0; r < 8; ++r) inv[r] = 1.0f / l[r];
#pragma unroll
  for (int dt = 0; dt < 8; ++dt)
#pragma unroll
    for (int r = 0; r < 8; ++r)
      __builtin_nontemporal_store(
          O[dt][r] * inv[r],
          out + (size_t)(qb + r + 8 * hl) * DIM + dt * 16 + c16);
}

// ---- launcher ------------------------------------------------------------

extern "C" void kernel_launch(void* const* d_in, const int* in_sizes, int n_in,
                              void* d_out, int out_size, void* d_ws, size_t ws_size,
                              hipStream_t stream) {
  const float* X  = (const float*)d_in[0];
  const float* Wq = (const float*)d_in[1];
  const float* bq = (const float*)d_in[2];
  const float* Wk = (const float*)d_in[3];
  const float* bk = (const float*)d_in[4];
  const float* Wv = (const float*)d_in[5];
  const float* bv = (const float*)d_in[6];
  const int*  adj = (const int*)d_in[7];

  _Float16* Qs = (_Float16*)d_ws;                      // 4 MB
  _Float16* Kh = Qs + (size_t)NROWS * DIM;             // 4 MB
  _Float16* Vt = Kh + (size_t)NROWS * DIM;             // 4 MB (transposed)

  qkv_kernel<<<dim3(NROWS / 128), dim3(256), 0, stream>>>(
      X, Wq, bq, Wk, bk, Wv, bv, Qs, Kh, Vt);
  attn_kernel<<<dim3(NROWS / 128), dim3(256), 0, stream>>>(
      Qs, Kh, Vt, adj, (float*)d_out);
}
